// HunyuanMoE_86775519248871
// MI455X (gfx1250) — matrix operand order repaired and checked by reading, not compiled, not run on hardware
//
#include <hip/hip_runtime.h>
#include <hip/hip_bf16.h>
#include <math.h>

#define S_TOK  1024
#define HID    768
#define NEXP   64
#define CAP    256
#define INTER  3072
#define INTER2 6144
#define TOPK   8
#define KC     128   // K-chunk staged in LDS
#define MT     64    // M-tile per block

typedef __attribute__((ext_vector_type(16))) __bf16 v16bf;
typedef __attribute__((ext_vector_type(8)))  float  v8f;

union Frag {
  v16bf bf;
  unsigned short us[16];
  uint4 q[2];
};

// native cast -> backend can use hardware cvt (pairs into v_cvt_pk_bf16_f32 where available)
__device__ __forceinline__ unsigned short f2bf(float f) {
  union { __bf16 h; unsigned short u; } c;
  c.h = (__bf16)f;
  return c.u;
}

__device__ __forceinline__ float siluf(float x) {
  return x / (1.0f + __expf(-x));
}

// ---- CDNA5 async copy: global -> LDS, tracked by ASYNCcnt ----
__device__ __forceinline__ void async_ld_b128(unsigned lds_off, unsigned long long gaddr) {
  asm volatile("global_load_async_to_lds_b128 %0, %1, off"
               :: "v"(lds_off), "v"(gaddr) : "memory");
}
__device__ __forceinline__ void wait_async0() {
  asm volatile("s_wait_asynccnt 0x0" ::: "memory");
}

// A-frag (bf16 16x32): lane row = lane&15, K-half = lane>>4.
// us[0..15] = K = ks + kh*16 + 0..15
__device__ __forceinline__ void load_a(const unsigned short* As, int row, int ks, int kh, Frag& a) {
  const unsigned short* base = As + row * KC + ks + kh * 16;
  a.q[0] = *(const uint4*)(base);
  a.q[1] = *(const uint4*)(base + 8);
}

// B-frag (bf16 32x16): lane col = lane&15, K-half = lane>>4.
// us[0..15] = K = kglob + 0..15 (kglob already includes kh*16), fp32 -> bf16 convert.
__device__ __forceinline__ void load_b(const float* wrow, int kglob, Frag& b) {
  const float4* p = (const float4*)(wrow + kglob);
#pragma unroll
  for (int i = 0; i < 4; ++i) {
    float4 v = p[i];
    b.bf[i * 4 + 0] = (__bf16)v.x;
    b.bf[i * 4 + 1] = (__bf16)v.y;
    b.bf[i * 4 + 2] = (__bf16)v.z;
    b.bf[i * 4 + 3] = (__bf16)v.w;
  }
}

// ---------------- elementwise fp32 -> bf16 ----------------
__global__ void k_convert_bf16(const float* __restrict__ x, unsigned short* __restrict__ o, int n) {
  int i = blockIdx.x * blockDim.x + threadIdx.x;
  if (i < n) o[i] = f2bf(x[i]);
}

// ---------------- router: logits -> softmax -> top8 (weights renormalized over top-k) ----------------
__global__ void k_router(const float* __restrict__ x, const float* __restrict__ wg,
                         int* __restrict__ tki, float* __restrict__ tkw) {
  __shared__ float xs[HID];
  __shared__ float p[NEXP];
  int s = blockIdx.x;
  int t = threadIdx.x;
  for (int h = t; h < HID; h += blockDim.x) xs[h] = x[s * HID + h];
  __syncthreads();
  if (t < NEXP) {
    const float* w = wg + t * HID;
    float acc = 0.f;
#pragma unroll 8
    for (int h = 0; h < HID; ++h) acc = fmaf(xs[h], w[h], acc);
    p[t] = acc;
  }
  __syncthreads();
  if (t == 0) {
    float mx = -3.4e38f;
    for (int e = 0; e < NEXP; ++e) mx = fmaxf(mx, p[e]);
    for (int e = 0; e < NEXP; ++e) p[e] = __expf(p[e] - mx);
    float tv[TOPK]; int ti[TOPK]; float ssum = 0.f;
    for (int k = 0; k < TOPK; ++k) {
      float bv = -1.f; int bi = 0;
      for (int e = 0; e < NEXP; ++e) if (p[e] > bv) { bv = p[e]; bi = e; }
      tv[k] = bv; ti[k] = bi; p[bi] = -2.f; ssum += bv;
    }
    float inv = 1.f / fmaxf(ssum, 1.1754944e-38f);
    for (int k = 0; k < TOPK; ++k) { tki[s * TOPK + k] = ti[k]; tkw[s * TOPK + k] = tv[k] * inv; }
  }
}

// ---------------- deterministic expert-major slot assignment ----------------
__global__ void k_assign(const int* __restrict__ tki, int* __restrict__ slot_assign,
                         int* __restrict__ token_of, int* __restrict__ cnt) {
  __shared__ int pre[257];
  int e = blockIdx.x;
  int t = threadIdx.x;
  const int CH = (TOPK * S_TOK) / 256;  // 32 flattened entries per thread, order f = kk*S + s
  int f0 = t * CH;
  int c = 0;
  for (int j = 0; j < CH; ++j) {
    int f = f0 + j;
    int kk = f >> 10; int s = f & (S_TOK - 1);
    if (tki[s * TOPK + kk] == e) c++;
  }
  pre[t] = c;
  __syncthreads();
  if (t == 0) {
    int run = 0;
    for (int i = 0; i < 256; ++i) { int v = pre[i]; pre[i] = run; run += v; }
    pre[256] = run;
    cnt[e] = run < CAP ? run : CAP;
  }
  __syncthreads();
  int slot = pre[t];
  for (int j = 0; j < CH; ++j) {
    int f = f0 + j;
    int kk = f >> 10; int s = f & (S_TOK - 1);
    if (tki[s * TOPK + kk] == e) {
      if (slot < CAP) { token_of[e * CAP + slot] = s; slot_assign[s * TOPK + kk] = slot; }
      else            { slot_assign[s * TOPK + kk] = -1; }
      slot++;
    }
  }
}

// ---------------- gather tokens into [E, CAP, HID] bf16 (zero-pad empty slots) ----------------
__global__ void k_gather(const unsigned short* __restrict__ xbf, const int* __restrict__ token_of,
                         const int* __restrict__ cnt, unsigned short* __restrict__ Xd) {
  int ec = blockIdx.x;
  int e = ec >> 8; int c = ec & (CAP - 1);
  int t = threadIdx.x;
  if (c < cnt[e]) {
    int s = token_of[ec];
    for (int h = t; h < HID; h += blockDim.x) Xd[(size_t)ec * HID + h] = xbf[s * HID + h];
  } else {
    for (int h = t; h < HID; h += blockDim.x) Xd[(size_t)ec * HID + h] = 0;
  }
}

// ---------------- GEMM1 + SwiGLU: X[M,768] @ W[6144,768]^T -> swiglu -> bf16 [M,3072] ----------------
// Block: 8 waves; wave tile = 64(M) x 16(N); grid = (3072/128, M/64, batches)
// A staged in LDS via double-buffered global_load_async_to_lds_b128 (ASYNCcnt).
__global__ void __launch_bounds__(256) k_gemm_swiglu(
    const unsigned short* __restrict__ Xb, long long xStride,
    const float* __restrict__ Wb, long long wStride,
    unsigned short* __restrict__ Ob, long long oStride) {
  __shared__ __align__(16) unsigned short As[2 * MT * KC];
  const int z = blockIdx.z;
  const unsigned short* X = Xb + (size_t)((long long)z * xStride);
  const float* W = Wb + (size_t)((long long)z * wStride);
  unsigned short* O = Ob + (size_t)((long long)z * oStride);

  const int t = threadIdx.x;
  const int wave = t >> 5;
  const int lane = t & 31;
  const int n  = lane & 15;
  const int kh = lane >> 4;
  const int mBase = blockIdx.y * MT;
  const int ncol  = blockIdx.x * 128 + wave * 16;

  v8f acc1[4] = {{}, {}, {}, {}};
  v8f acc2[4] = {{}, {}, {}, {}};

  const float* w1row = W + (size_t)(ncol + n) * HID;          // up half
  const float* w2row = W + (size_t)(ncol + n + INTER) * HID;  // gate half (SiLU)

  const int srow = t >> 2;          // 64 rows, 4 threads per row
  const int sq   = (t & 3) * 32;    // 32 bf16 (64B) per thread

  // issue async stage of one 64x128 chunk into buffer `buf`
  auto stage = [&](int buf, int kc) {
    unsigned long long g = (unsigned long long)(X + (size_t)(mBase + srow) * HID + kc + sq);
    unsigned lds = (unsigned)(unsigned long long)(As + buf * (MT * KC) + srow * KC + sq);
#pragma unroll
    for (int i = 0; i < 4; ++i) async_ld_b128(lds + 16u * i, g + 32ull * i);
  };

  stage(0, 0);
  wait_async0();
  __syncthreads();

  int cur = 0;
  for (int kc = 0; kc < HID; kc += KC) {
    if (kc + KC < HID) {
      stage(cur ^ 1, kc + KC);                 // overlap next-chunk DMA with compute
      __builtin_prefetch(w1row + kc + KC, 0, 1);
      __builtin_prefetch(w2row + kc + KC, 0, 1);
    }
    const unsigned short* Ab = As + cur * (MT * KC);
#pragma unroll
    for (int ks = 0; ks < KC; ks += 32) {
      Frag b1, b2;
      const int kglob = kc + ks + kh * 16;
      load_b(w1row, kglob, b1);
      load_b(w2row, kglob, b2);
#pragma unroll
      for (int sub = 0; sub < 4; ++sub) {
        Frag a;
        load_a(Ab, sub * 16 + n, ks, kh, a);
        acc1[sub] = __builtin_amdgcn_wmma_f32_16x16x32_bf16(false, a.bf, false, b1.bf, (short)0, acc1[sub], false, false);
        acc2[sub] = __builtin_amdgcn_wmma_f32_16x16x32_bf16(false, a.bf, false, b2.bf, (short)0, acc2[sub], false, false);
      }
    }
    wait_async0();
    __syncthreads();
    cur ^= 1;
  }

  const int col = ncol + n;
#pragma unroll
  for (int sub = 0; sub < 4; ++sub) {
#pragma unroll
    for (int r = 0; r < 8; ++r) {
      int row = mBase + sub * 16 + kh * 8 + r;
      O[(size_t)row * INTER + col] = f2bf(acc1[sub][r] * siluf(acc2[sub][r]));
    }
  }
}

// ---------------- GEMM2 (down): G[M,3072] @ W[768,3072]^T -> fp32 [M,768] ----------------
// grid = (768/128, M/64, batches)
__global__ void __launch_bounds__(256) k_gemm_down(
    const unsigned short* __restrict__ Gb, long long gStride,
    const float* __restrict__ Wb, long long wStride,
    float* __restrict__ Ob, long long oStride) {
  __shared__ __align__(16) unsigned short As[2 * MT * KC];
  const int z = blockIdx.z;
  const unsigned short* G = Gb + (size_t)((long long)z * gStride);
  const float* W = Wb + (size_t)((long long)z * wStride);
  float* O = Ob + (size_t)((long long)z * oStride);

  const int t = threadIdx.x;
  const int wave = t >> 5;
  const int lane = t & 31;
  const int n  = lane & 15;
  const int kh = lane >> 4;
  const int mBase = blockIdx.y * MT;
  const int ncol  = blockIdx.x * 128 + wave * 16;

  v8f acc[4] = {{}, {}, {}, {}};
  const float* wrow = W + (size_t)(ncol + n) * INTER;

  const int srow = t >> 2;
  const int sq   = (t & 3) * 32;

  auto stage = [&](int buf, int kc) {
    unsigned long long g = (unsigned long long)(G + (size_t)(mBase + srow) * INTER + kc + sq);
    unsigned lds = (unsigned)(unsigned long long)(As + buf * (MT * KC) + srow * KC + sq);
#pragma unroll
    for (int i = 0; i < 4; ++i) async_ld_b128(lds + 16u * i, g + 32ull * i);
  };

  stage(0, 0);
  wait_async0();
  __syncthreads();

  int cur = 0;
  for (int kc = 0; kc < INTER; kc += KC) {
    if (kc + KC < INTER) {
      stage(cur ^ 1, kc + KC);
      __builtin_prefetch(wrow + kc + KC, 0, 1);
    }
    const unsigned short* Ab = As + cur * (MT * KC);
#pragma unroll
    for (int ks = 0; ks < KC; ks += 32) {
      Frag b;
      load_b(wrow, kc + ks + kh * 16, b);
#pragma unroll
      for (int sub = 0; sub < 4; ++sub) {
        Frag a;
        load_a(Ab, sub * 16 + n, ks, kh, a);
        acc[sub] = __builtin_amdgcn_wmma_f32_16x16x32_bf16(false, a.bf, false, b.bf, (short)0, acc[sub], false, false);
      }
    }
    wait_async0();
    __syncthreads();
    cur ^= 1;
  }

  const int col = ncol + n;
#pragma unroll
  for (int sub = 0; sub < 4; ++sub) {
#pragma unroll
    for (int r = 0; r < 8; ++r) {
      int row = mBase + sub * 16 + kh * 8 + r;
      O[(size_t)row * HID + col] = acc[sub][r];
    }
  }
}

// ---------------- combine: out = shared + sum_k w_k * expert_out[e_k, slot_k] ----------------
__global__ void k_combine(const float* __restrict__ so, const float* __restrict__ eo,
                          const int* __restrict__ tki, const float* __restrict__ tkw,
                          const int* __restrict__ slot_assign, float* __restrict__ out) {
  int s = blockIdx.x; int t = threadIdx.x;
  int ebuf[TOPK]; int sbuf[TOPK]; float wbuf[TOPK];
#pragma unroll
  for (int k = 0; k < TOPK; ++k) {
    ebuf[k] = tki[s * TOPK + k];
    sbuf[k] = slot_assign[s * TOPK + k];
    wbuf[k] = tkw[s * TOPK + k];
  }
  for (int h = t; h < HID; h += blockDim.x) {
    float acc = so[(size_t)s * HID + h];
#pragma unroll
    for (int k = 0; k < TOPK; ++k) {
      int sl = sbuf[k];
      if (sl >= 0) acc += wbuf[k] * eo[((size_t)(ebuf[k] * CAP + sl)) * HID + h];
    }
    out[(size_t)s * HID + h] = acc;
  }
}

extern "C" void kernel_launch(void* const* d_in, const int* in_sizes, int n_in,
                              void* d_out, int out_size, void* d_ws, size_t ws_size,
                              hipStream_t stream) {
  (void)in_sizes; (void)n_in; (void)out_size; (void)ws_size;
  const float* hs     = (const float*)d_in[0];  // [1,1024,768]
  const float* w_gu   = (const float*)d_in[1];  // [6144,768]
  const float* w_down = (const float*)d_in[2];  // [768,3072]
  const float* wg     = (const float*)d_in[3];  // [64,768]
  const float* w1     = (const float*)d_in[4];  // [64,6144,768]
  const float* w2     = (const float*)d_in[5];  // [64,768,3072]
  float* out = (float*)d_out;

  char* ws = (char*)d_ws;
  size_t off = 0;
  auto alloc = [&](size_t bytes) -> void* {
    void* p = ws + off;
    off = (off + bytes + 255) & ~(size_t)255;
    return p;
  };

  unsigned short* xbf   = (unsigned short*)alloc((size_t)S_TOK * HID * 2);
  int*   tki            = (int*)  alloc((size_t)S_TOK * TOPK * 4);
  float* tkw            = (float*)alloc((size_t)S_TOK * TOPK * 4);
  int*   slot_assign    = (int*)  alloc((size_t)S_TOK * TOPK * 4);
  int*   token_of       = (int*)  alloc((size_t)NEXP * CAP * 4);
  int*   cnt            = (int*)  alloc((size_t)NEXP * 4);
  unsigned short* Xd    = (unsigned short*)alloc((size_t)NEXP * CAP * HID * 2);
  unsigned short* Gated = (unsigned short*)alloc((size_t)NEXP * CAP * INTER * 2);
  float* eo             = (float*)alloc((size_t)NEXP * CAP * HID * 4);
  unsigned short* sg    = (unsigned short*)alloc((size_t)S_TOK * INTER * 2);
  float* so             = (float*)alloc((size_t)S_TOK * HID * 4);

  const int nconv = S_TOK * HID;
  k_convert_bf16<<<(nconv + 255) / 256, 256, 0, stream>>>(hs, xbf, nconv);
  k_router<<<S_TOK, 256, 0, stream>>>(hs, wg, tki, tkw);
  k_assign<<<NEXP, 256, 0, stream>>>(tki, slot_assign, token_of, cnt);
  k_gather<<<NEXP * CAP, 128, 0, stream>>>(xbf, token_of, cnt, Xd);

  // shared MLP (batch = 1, M = 1024)
  dim3 g1s(INTER / 128, S_TOK / MT, 1);
  k_gemm_swiglu<<<g1s, 256, 0, stream>>>(xbf, 0, w_gu, 0, sg, 0);
  dim3 g2s(HID / 128, S_TOK / MT, 1);
  k_gemm_down<<<g2s, 256, 0, stream>>>(sg, 0, w_down, 0, so, 0);

  // experts (batch = 64, M = 256)
  dim3 g1e(INTER / 128, CAP / MT, NEXP);
  k_gemm_swiglu<<<g1e, 256, 0, stream>>>(Xd, (long long)CAP * HID,
                                         w1, (long long)INTER2 * HID,
                                         Gated, (long long)CAP * INTER);
  dim3 g2e(HID / 128, CAP / MT, NEXP);
  k_gemm_down<<<g2e, 256, 0, stream>>>(Gated, (long long)CAP * INTER,
                                       w2, (long long)HID * INTER,
                                       eo, (long long)CAP * HID);

  k_combine<<<S_TOK, 256, 0, stream>>>(so, eo, tki, tkw, slot_assign, out);
}
